// MultiheadAttention_45492293599551
// MI455X (gfx1250) — matrix operand order repaired and checked by reading, not compiled, not run on hardware
//
#include <hip/hip_runtime.h>
#include <hip/hip_bf16.h>

#define S_LEN 2048
#define EMB   1024
#define HEADS 16
#define HD    64
#define MROWS 4096  // B * S

typedef __attribute__((ext_vector_type(16))) __bf16 v16bf;
typedef __attribute__((ext_vector_type(8)))  float  v8f;

union BF16Frag {
  uint4          u4[2];
  unsigned short h[16];
  v16bf          v;
};

__device__ __forceinline__ unsigned short f32_to_bf16(float f) {
  union { float f; unsigned int u; } x;
  x.f = f;
  unsigned int r = x.u + 0x7FFFu + ((x.u >> 16) & 1u);  // round-to-nearest-even
  return (unsigned short)(r >> 16);
}

__device__ __forceinline__ v8f wmma_bf16(v16bf a, v16bf b, v8f c) {
  // (neg_a, A, neg_b, B, c_mod, C, reuse_a, reuse_b)
  return __builtin_amdgcn_wmma_f32_16x16x32_bf16(false, a, false, b, (short)0, c,
                                                 false, false);
}

// ---------------------------------------------------------------------------
// Kernel 0: one-shot f32 -> bf16 conversion (memory bound; hides the cvt ALU
// that previously ran 64x redundantly inside the GEMM inner loop).
// ---------------------------------------------------------------------------
__global__ __launch_bounds__(256)
void cvt_bf16_kernel(const float* __restrict__ src,
                     unsigned short* __restrict__ dst, int n4) {
  const int i = blockIdx.x * 256 + threadIdx.x;
  if (i >= n4) return;
  const float4 f = ((const float4*)src)[i];
  ushort4 o;
  o.x = f32_to_bf16(f.x);
  o.y = f32_to_bf16(f.y);
  o.z = f32_to_bf16(f.z);
  o.w = f32_to_bf16(f.w);
  ((ushort4*)dst)[i] = o;
}

// ---------------------------------------------------------------------------
// Kernel 1: QKV projection, all-bf16 operands.  P = x @ W^T + bias.
// blockIdx.z selects Q/K/V.  Q,K -> bf16 [B,H,S,D]; V -> bf16 [B,H,D,S].
// block = 128 = 4 waves; wave computes a 32(M) x 64(N) strip (8 WMMA / K-step).
// ---------------------------------------------------------------------------
__global__ __launch_bounds__(128)
void qkv_proj_kernel(const unsigned short* __restrict__ xb,   // 3 x [4096,1024]
                     const unsigned short* __restrict__ Wb,   // 3 x [1024,1024]
                     const float* __restrict__ bq, const float* __restrict__ bk,
                     const float* __restrict__ bv,
                     unsigned short* __restrict__ Qh,
                     unsigned short* __restrict__ Kh,
                     unsigned short* __restrict__ Vt) {
  const int which = blockIdx.z;
  const unsigned short* x = xb + (size_t)which * MROWS * EMB;
  const unsigned short* W = Wb + (size_t)which * EMB * EMB;
  const float* bias = (which == 0) ? bq : (which == 1) ? bk : bv;

  const int lane = threadIdx.x & 31;
  const int wave = threadIdx.x >> 5;
  const int hf   = lane >> 4;
  const int l16  = lane & 15;

  const int m0 = (blockIdx.x * 4 + wave) * 32;  // 32-row strip in [0,4096)
  const int n0 = blockIdx.y * 64;               // output-channel strip == head*64
  const int h  = blockIdx.y;

  v8f acc[2][4];
#pragma unroll
  for (int mt = 0; mt < 2; ++mt)
#pragma unroll
    for (int t = 0; t < 4; ++t)
#pragma unroll
      for (int r = 0; r < 8; ++r) acc[mt][t][r] = 0.0f;

  // per-lane base pointers (A layout: lanes 0-15 K{0-7,16-23}; 16-31 K{8-15,24-31})
  const unsigned short* ap0 = x + (size_t)(m0 + l16) * EMB + (hf ? 8 : 0);
  const unsigned short* ap1 = ap0 + (size_t)16 * EMB;
  const unsigned short* wp  = W + (size_t)(n0 + l16) * EMB + (hf ? 8 : 0);

  for (int k0 = 0; k0 < EMB; k0 += 32) {
    BF16Frag a0, a1;
    a0.u4[0] = *(const uint4*)(ap0 + k0);
    a0.u4[1] = *(const uint4*)(ap0 + k0 + 16);
    a1.u4[0] = *(const uint4*)(ap1 + k0);
    a1.u4[1] = *(const uint4*)(ap1 + k0 + 16);
#pragma unroll
    for (int t = 0; t < 4; ++t) {
      // B fragment: B[k,n] = W[n,k]; lane column n = n0+t*16+l16 -> W row,
      // same K order as A (lanes 0-15: K{0-7,16-23}; 16-31: K{8-15,24-31})
      BF16Frag b;
      const unsigned short* bp = wp + (size_t)(t * 16) * EMB + k0;
      b.u4[0] = *(const uint4*)(bp);
      b.u4[1] = *(const uint4*)(bp + 16);
      acc[0][t] = wmma_bf16(a0.v, b.v, acc[0][t]);
      acc[1][t] = wmma_bf16(a1.v, b.v, acc[1][t]);
    }
  }

  // bias + store (C layout: VGPR r, lane half -> row r + 8*hf, col l16)
#pragma unroll
  for (int t = 0; t < 4; ++t) {
    const int d   = t * 16 + l16;  // within-head dim
    const float bn = bias[n0 + d];
#pragma unroll
    for (int mt = 0; mt < 2; ++mt) {
#pragma unroll
      for (int r = 0; r < 8; ++r) {
        const int m = m0 + mt * 16 + r + 8 * hf;  // global row in [0,4096)
        const int b = m >> 11;                    // batch (S=2048)
        const int s = m & (S_LEN - 1);
        const unsigned short hv = f32_to_bf16(acc[mt][t][r] + bn);
        if (which == 2) {
          Vt[(((size_t)(b * HEADS + h) * HD) + d) * S_LEN + s] = hv;
        } else {
          unsigned short* dst = (which == 0) ? Qh : Kh;
          dst[((size_t)(b * HEADS + h) * S_LEN + s) * HD + d] = hv;
        }
      }
    }
  }
}

// ---------------------------------------------------------------------------
// Kernel 2: flash attention.  Per (b,h): O = softmax(Q K^T / 8) V.
// grid = (S/64, B*H); block = 128 = 4 waves; wave owns 16 query rows.
// ---------------------------------------------------------------------------
__global__ __launch_bounds__(128)
void attn_kernel(const unsigned short* __restrict__ Qh,
                 const unsigned short* __restrict__ Kh,
                 const unsigned short* __restrict__ Vt,
                 float* __restrict__ out) {
  __shared__ __align__(16) unsigned short Plds[4 * 16 * 56];

  const int lane = threadIdx.x & 31;
  const int wave = threadIdx.x >> 5;
  const int hf   = lane >> 4;
  const int l16  = lane & 15;
  const int bh   = blockIdx.y;                   // [0, B*H)
  const int q0   = blockIdx.x * 64 + wave * 16;  // query tile base

  const unsigned short* Qb = Qh + (size_t)bh * S_LEN * HD;
  const unsigned short* Kb = Kh + (size_t)bh * S_LEN * HD;
  const unsigned short* Vb = Vt + (size_t)bh * HD * S_LEN;

  // Q A-fragments for K-dim 0..31 and 32..63 (loaded once)
  BF16Frag qa[2];
  {
    const unsigned short* qp = Qb + (size_t)(q0 + l16) * HD + (hf ? 8 : 0);
#pragma unroll
    for (int f = 0; f < 2; ++f) {
      qa[f].u4[0] = *(const uint4*)(qp + f * 32);
      qa[f].u4[1] = *(const uint4*)(qp + f * 32 + 16);
    }
  }

  float rowM[8], rowL[8];
  v8f acc[4];
#pragma unroll
  for (int r = 0; r < 8; ++r) { rowM[r] = -INFINITY; rowL[r] = 0.0f; }
#pragma unroll
  for (int t = 0; t < 4; ++t)
#pragma unroll
    for (int r = 0; r < 8; ++r) acc[t][r] = 0.0f;

  unsigned short* pbase = Plds + wave * 16 * 56;
  const float scale = 0.125f;  // 1/sqrt(64)

  const unsigned short* kbase = Kb + (size_t)l16 * HD + (hf ? 8 : 0);
  const unsigned short* vbase = Vb + (size_t)l16 * S_LEN + (hf ? 8 : 0);

  for (int j0 = 0; j0 < S_LEN; j0 += 32) {
    // prefetch next key block's K/V lines into cache (global_prefetch_b8)
    if (j0 + 32 < S_LEN) {
      __builtin_prefetch(kbase + (size_t)(j0 + 32) * HD, 0, 1);
      __builtin_prefetch(vbase + (j0 + 32), 0, 1);
    }

    // ---- logits: two 16x16 tiles over keys j0..j0+31, contracted over D=64
    v8f s01[2];
#pragma unroll
    for (int tile = 0; tile < 2; ++tile) {
      const unsigned short* kp = kbase + (size_t)(j0 + tile * 16) * HD;
      BF16Frag kb0, kb1;
      kb0.u4[0] = *(const uint4*)(kp);      kb0.u4[1] = *(const uint4*)(kp + 16);
      kb1.u4[0] = *(const uint4*)(kp + 32); kb1.u4[1] = *(const uint4*)(kp + 48);
      v8f sc;
#pragma unroll
      for (int r = 0; r < 8; ++r) sc[r] = 0.0f;
      sc = wmma_bf16(qa[0].v, kb0.v, sc);
      sc = wmma_bf16(qa[1].v, kb1.v, sc);
      s01[tile] = sc;
    }

    // ---- online softmax update (row = r + 8*hf; 16 cols live across a lane-half)
    float alpha[8], p0[8], p1[8];
#pragma unroll
    for (int r = 0; r < 8; ++r) {
      const float a0 = s01[0][r] * scale;
      const float a1 = s01[1][r] * scale;
      float mj = fmaxf(a0, a1);
#pragma unroll
      for (int off = 1; off < 16; off <<= 1)
        mj = fmaxf(mj, __shfl_xor(mj, off, 32));
      const float Mn = fmaxf(rowM[r], mj);
      alpha[r] = __expf(rowM[r] - Mn);
      rowM[r]  = Mn;
      p0[r] = __expf(a0 - Mn);
      p1[r] = __expf(a1 - Mn);
      float rs = p0[r] + p1[r];
#pragma unroll
      for (int off = 1; off < 16; off <<= 1) rs += __shfl_xor(rs, off, 32);
      rowL[r] = rowL[r] * alpha[r] + rs;
    }
#pragma unroll
    for (int t = 0; t < 4; ++t)
#pragma unroll
      for (int r = 0; r < 8; ++r) acc[t][r] *= alpha[r];

    // ---- C-layout -> A-layout via per-wave LDS tile (16 x 32, stride 56)
#pragma unroll
    for (int r = 0; r < 8; ++r) {
      const int m = r + 8 * hf;
      pbase[m * 56 + l16]      = f32_to_bf16(p0[r]);
      pbase[m * 56 + 16 + l16] = f32_to_bf16(p1[r]);
    }
    BF16Frag pf;
    {
      const unsigned short* pp = pbase + l16 * 56 + (hf ? 8 : 0);
      pf.u4[0] = *(const uint4*)(pp);
      pf.u4[1] = *(const uint4*)(pp + 16);
    }

    // ---- O += P @ V ; B-fragment from transposed V (rows = d, contiguous keys)
#pragma unroll
    for (int t = 0; t < 4; ++t) {
      const unsigned short* vp = vbase + (size_t)(t * 16) * S_LEN + j0;
      BF16Frag vbf;
      vbf.u4[0] = *(const uint4*)(vp);
      vbf.u4[1] = *(const uint4*)(vp + 16);
      acc[t] = wmma_bf16(pf.v, vbf.v, acc[t]);
    }
  }

  // ---- epilogue: normalize and store f32 [B,S,E]
  const int b = bh >> 4;
  const int h = bh & 15;
#pragma unroll
  for (int t = 0; t < 4; ++t) {
    const int d = t * 16 + l16;
#pragma unroll
    for (int r = 0; r < 8; ++r) {
      const int s = q0 + r + 8 * hf;
      out[((size_t)(b * S_LEN + s)) * EMB + h * HD + d] = acc[t][r] / rowL[r];
    }
  }
}

// ---------------------------------------------------------------------------
extern "C" void kernel_launch(void* const* d_in, const int* in_sizes, int n_in,
                              void* d_out, int out_size, void* d_ws, size_t ws_size,
                              hipStream_t stream) {
  const float* xin[3] = {(const float*)d_in[0], (const float*)d_in[1],
                         (const float*)d_in[2]};
  const float* Win[3] = {(const float*)d_in[3], (const float*)d_in[5],
                         (const float*)d_in[7]};
  const float* bq = (const float*)d_in[4];
  const float* bk = (const float*)d_in[6];
  const float* bv = (const float*)d_in[8];

  // bf16 workspace layout (ushort elements):
  //   Qh, Kh  [B,H,S,D]      : 4,194,304 each
  //   Vt      [B,H,D,S]      : 4,194,304
  //   xb      3 x [4096,1024]: 4,194,304 each
  //   Wb      3 x [1024,1024]: 1,048,576 each
  const size_t eQ = (size_t)2 * HEADS * S_LEN * HD;  // 4 Mi
  const size_t eX = (size_t)MROWS * EMB;             // 4 Mi
  const size_t eW = (size_t)EMB * EMB;               // 1 Mi
  unsigned short* Qh = (unsigned short*)d_ws;
  unsigned short* Kh = Qh + eQ;
  unsigned short* Vt = Kh + eQ;
  unsigned short* xb = Vt + eQ;
  unsigned short* Wb = xb + 3 * eX;

  // one-shot f32 -> bf16 conversions (x3 activations, x3 weights)
  for (int i = 0; i < 3; ++i) {
    const int nx4 = (int)(eX / 4);
    const int nw4 = (int)(eW / 4);
    cvt_bf16_kernel<<<(nx4 + 255) / 256, 256, 0, stream>>>(xin[i], xb + i * eX, nx4);
    cvt_bf16_kernel<<<(nw4 + 255) / 256, 256, 0, stream>>>(Win[i], Wb + i * eW, nw4);
  }

  qkv_proj_kernel<<<dim3(MROWS / 128, 16, 3), 128, 0, stream>>>(
      xb, Wb, bq, bk, bv, Qh, Kh, Vt);
  attn_kernel<<<dim3(S_LEN / 64, 2 * HEADS), 128, 0, stream>>>(
      Qh, Kh, Vt, (float*)d_out);
}
